// WindowAttention_78426102825239
// MI455X (gfx1250) — matrix operand order repaired and checked by reading, not compiled, not run on hardware
//
#include <hip/hip_runtime.h>

typedef __bf16 bf16_t;
typedef bf16_t v16bf __attribute__((ext_vector_type(16)));
typedef bf16_t v8bf  __attribute__((ext_vector_type(8)));
typedef float  v8f   __attribute__((ext_vector_type(8)));

#define DIM   512
#define HEADS 8
#define HD    64
#define BB    128
#define NTOK  256
#define MROWS (BB * NTOK)   // 32768 tokens

// ---------------------------------------------------------------- WMMA helpers

__device__ __forceinline__ v8f wmma_bf16(v16bf a, v16bf b, v8f c) {
  // D = A(16x32) * B(32x16) + C(16x16), f32 accumulate
  return __builtin_amdgcn_wmma_f32_16x16x32_bf16(false, a, false, b, (short)0, c,
                                                 false, false);
}

// A-matrix fragment, 16-bit 16x32: lane = {khi,m}; within a lane the
// 16 halves are contiguous K = 16*khi .. 16*khi + 15 (matches frag_b).
__device__ __forceinline__ v16bf frag_a(const bf16_t* p, int khi) {
  v8bf lo = *(const v8bf*)(p + 16 * khi);
  v8bf hi = *(const v8bf*)(p + 16 * khi + 8);
  v16bf r;
#pragma unroll
  for (int i = 0; i < 8; ++i) { r[i] = lo[i]; r[i + 8] = hi[i]; }
  return r;
}

// B-matrix fragment (32x16, K-major rows): lane n = lane%16, lanes 16-31 take the
// upper K half; caller bakes the +16*khi into p, so 16 contiguous halves here.
__device__ __forceinline__ v16bf frag_b(const bf16_t* p) {
  v8bf lo = *(const v8bf*)(p);
  v8bf hi = *(const v8bf*)(p + 8);
  v16bf r;
#pragma unroll
  for (int i = 0; i < 8; ++i) { r[i] = lo[i]; r[i + 8] = hi[i]; }
  return r;
}

__device__ __forceinline__ float uload(const float* p) {  // pin uniform to SGPR
  return __int_as_float(__builtin_amdgcn_readfirstlane(__float_as_int(*p)));
}

// ---------------------------------------------------------------- small kernels

__global__ void k_f32_to_bf16(const float* __restrict__ src,
                              bf16_t* __restrict__ dst, int n) {
  int i = blockIdx.x * blockDim.x + threadIdx.x;
  int stride = gridDim.x * blockDim.x;
  for (; i < n; i += stride) dst[i] = (bf16_t)src[i];
}

// vote-embed MLP: (B*N,3) -> relu(16) -> (512), bf16 out, one token per block
__global__ void k_vote_mlp(const float* __restrict__ vote,
                           const float* __restrict__ w1, const float* __restrict__ b1,
                           const float* __restrict__ w2, const float* __restrict__ b2,
                           bf16_t* __restrict__ veb) {
  const int tok = blockIdx.x;
  const float v0 = vote[tok * 3 + 0];
  const float v1 = vote[tok * 3 + 1];
  const float v2 = vote[tok * 3 + 2];
  float h[16];
#pragma unroll
  for (int o = 0; o < 16; ++o)
    h[o] = fmaxf(w1[o * 3 + 0] * v0 + w1[o * 3 + 1] * v1 + w1[o * 3 + 2] * v2 +
                 b1[o], 0.f);
#pragma unroll
  for (int rep = 0; rep < 2; ++rep) {
    const int c = threadIdx.x + rep * 256;
    const float* wr = w2 + (size_t)c * 16;
    float acc = b2[c];
#pragma unroll
    for (int o = 0; o < 16; ++o) acc += wr[o] * h[o];
    veb[(size_t)tok * DIM + c] = (bf16_t)acc;
  }
}

// q/k row L2 norms; one row per wave (wave32)
__global__ void k_norms(const bf16_t* __restrict__ qb, const bf16_t* __restrict__ kb,
                        float* __restrict__ qn, float* __restrict__ kn) {
  const int lane = threadIdx.x & 31;
  const size_t row = (size_t)blockIdx.x * 8 + (threadIdx.x >> 5);
  {
    const bf16_t* p = qb + row * HD + lane * 2;
    float a = (float)p[0], b = (float)p[1];
    float s = a * a + b * b;
#pragma unroll
    for (int off = 16; off >= 1; off >>= 1) s += __shfl_xor(s, off, 32);
    if (lane == 0) qn[row] = sqrtf(s);
  }
  {
    const bf16_t* p = kb + row * HD + lane * 2;
    float a = (float)p[0], b = (float)p[1];
    float s = a * a + b * b;
#pragma unroll
    for (int off = 16; off >= 1; off >>= 1) s += __shfl_xor(s, off, 32);
    if (lane == 0) kn[row] = sqrtf(s);
  }
}

// ------------------------------------------------- QKV GEMM + ve fuse (WMMA bf16)
// C[m,j] = sum_k x[m,k]*Wqkv[j,k]; wave tile 16x64, block = 8 waves (M=128)

__global__ __launch_bounds__(256) void
k_qkv(const bf16_t* __restrict__ xb, const bf16_t* __restrict__ wb,
      const bf16_t* __restrict__ veb, bf16_t* __restrict__ qb,
      bf16_t* __restrict__ kbuf, bf16_t* __restrict__ vbuf) {
  const int lane = threadIdx.x & 31, w = threadIdx.x >> 5;
  const int ln = lane & 15, lh = lane >> 4;
  const int m0 = (blockIdx.y * 8 + w) * 16;
  const int j0 = blockIdx.x * 64;

  const bf16_t* arow  = xb + (size_t)(m0 + ln) * DIM;
  const bf16_t* brow0 = wb + (size_t)(j0 + ln) * DIM + 16 * lh;

  v8f acc[4];
#pragma unroll
  for (int t = 0; t < 4; ++t) acc[t] = (v8f){0.f,0.f,0.f,0.f,0.f,0.f,0.f,0.f};

  for (int kk = 0; kk < DIM; kk += 32) {
    __builtin_prefetch(arow + kk + 128, 0, 1);
    v16bf a = frag_a(arow + kk, lh);
#pragma unroll
    for (int t = 0; t < 4; ++t) {
      v16bf b = frag_b(brow0 + (size_t)t * 16 * DIM + kk);
      acc[t] = wmma_bf16(a, b, acc[t]);
    }
  }

  const int sec  = j0 / DIM;                 // 0=q 1=k 2=v (64-col tile stays inside)
  const int h    = (j0 % DIM) / HD;
  const int bidx = m0 / NTOK;
  const float vescale = (sec == 2) ? 2.f : 1.f;  // reference adds ve to v twice
  bf16_t* dst = (sec == 0) ? qb : (sec == 1) ? kbuf : vbuf;

#pragma unroll
  for (int t = 0; t < 4; ++t) {
#pragma unroll
    for (int r = 0; r < 8; ++r) {
      const int n = (m0 % NTOK) + r + 8 * lh;   // D-layout: VGPR r, lane half lh
      const int d = t * 16 + ln;
      const float ve = (float)veb[((size_t)(bidx * NTOK + n)) * DIM + h * HD + d];
      const float val = acc[t][r] + vescale * ve;
      dst[((size_t)((bidx * HEADS + h) * NTOK + n)) * HD + d] = (bf16_t)val;
    }
  }
}

// ------------------------------------------------------- attention per (b,h) block
// LDS: Q(32K) K(32K) V^T(32K) + norms/pos(4K) + per-wave P(64K) = 164 KB (<320 KB)

__global__ __launch_bounds__(256) void
k_attn(const bf16_t* __restrict__ qb, const bf16_t* __restrict__ kb,
       const bf16_t* __restrict__ vb, const float* __restrict__ qn,
       const float* __restrict__ kn, const float* __restrict__ pos,
       const float* __restrict__ tau, const float* __restrict__ rw1,
       const float* __restrict__ rb1, const float* __restrict__ rw2,
       const float* __restrict__ rb2, bf16_t* __restrict__ ao) {
  extern __shared__ char smem[];
  bf16_t* Qs   = (bf16_t*)smem;
  bf16_t* Ks   = Qs + NTOK * HD;
  bf16_t* VT   = Ks + NTOK * HD;
  float*  qn_s = (float*)(VT + NTOK * HD);
  float*  kn_s = qn_s + NTOK;
  float*  pxs  = kn_s + NTOK;
  float*  pys  = pxs + NTOK;
  bf16_t* Pall = (bf16_t*)(pys + NTOK);

  const int tid = threadIdx.x;
  const int lane = tid & 31, w = tid >> 5;
  const int ln = lane & 15, lh = lane >> 4;
  const int bh = blockIdx.x;
  const int b = bh / HEADS, h = bh % HEADS;
  const size_t base = (size_t)bh * NTOK * HD;

  // cooperative stage: Q, K row-major; V transposed (d-major) for the PV B-operand
  {
    const uint4* qsrc = (const uint4*)(qb + base);
    const uint4* ksrc = (const uint4*)(kb + base);
    uint4* qdst = (uint4*)Qs;
    uint4* kdst = (uint4*)Ks;
    for (int i = tid; i < NTOK * HD / 8; i += 256) { qdst[i] = qsrc[i]; kdst[i] = ksrc[i]; }
    const uint4* vsrc = (const uint4*)(vb + base);
    for (int i = tid; i < NTOK * HD / 8; i += 256) {
      uint4 u = vsrc[i];
      const bf16_t* e = (const bf16_t*)&u;
      const int j = i >> 3;            // token row (8 x uint4 per 64-wide row)
      const int d0 = (i & 7) * 8;
#pragma unroll
      for (int q = 0; q < 8; ++q) VT[(d0 + q) * NTOK + j] = e[q];
    }
    for (int i = tid; i < NTOK; i += 256) {
      qn_s[i] = qn[(size_t)bh * NTOK + i];
      kn_s[i] = kn[(size_t)bh * NTOK + i];
      pxs[i]  = pos[((size_t)b * NTOK + i) * 2 + 0];
      pys[i]  = pos[((size_t)b * NTOK + i) * 2 + 1];
    }
  }
  __syncthreads();

  // RPE weights pinned to SGPRs (wave-uniform)
  float sw1x[16], sw1y[16], sb1[16], sw2[16];
#pragma unroll
  for (int o = 0; o < 16; ++o) {
    sw1x[o] = uload(rw1 + o * 2 + 0);
    sw1y[o] = uload(rw1 + o * 2 + 1);
    sb1[o]  = uload(rb1 + o);
    sw2[o]  = uload(rw2 + h * 16 + o);
  }
  const float sb2     = uload(rb2 + h);
  const float inv_tau = 1.f / fmaxf(uload(tau + h), 0.01f);
  const float scale   = 0.125f;  // 64^-0.5

  bf16_t* Pw = Pall + (size_t)w * 16 * NTOK;

  for (int it = 0; it < 2; ++it) {
    const int m0 = (w * 2 + it) * 16;

    // ---- S = Q K^T (16 x 256 strip), 32 WMMAs
    v8f S[16];
#pragma unroll
    for (int t = 0; t < 16; ++t) S[t] = (v8f){0.f,0.f,0.f,0.f,0.f,0.f,0.f,0.f};
    const bf16_t* qrow = Qs + (size_t)(m0 + ln) * HD;
#pragma unroll
    for (int kk = 0; kk < HD; kk += 32) {
      v16bf a = frag_a(qrow + kk, lh);
#pragma unroll
      for (int t = 0; t < 16; ++t) {
        v16bf bmat = frag_b(Ks + (size_t)(t * 16 + ln) * HD + kk + 16 * lh);
        S[t] = wmma_bf16(a, bmat, S[t]);
      }
    }

    // ---- cosine norm / temperature / RPE (f32 VALU), rows owned by (r, lh)
    float pix[8], piy[8], qnr[8], rmax[8], rsum[8];
#pragma unroll
    for (int r = 0; r < 8; ++r) {
      const int i = m0 + r + 8 * lh;
      pix[r] = pxs[i]; piy[r] = pys[i]; qnr[r] = qn_s[i];
      rmax[r] = -3.0e38f;
    }
#pragma unroll
    for (int t = 0; t < 16; ++t) {
      const int j = t * 16 + ln;
      const float knj = kn_s[j], pjx = pxs[j], pjy = pys[j];
#pragma unroll
      for (int r = 0; r < 8; ++r) {
        float s = S[t][r] * scale;
        s = s / fmaxf(qnr[r] * knj, 1e-6f);
        s *= inv_tau;
        const float dx = pix[r] - pjx, dy = piy[r] - pjy;
        float rp = sb2;
#pragma unroll
        for (int o = 0; o < 16; ++o)
          rp += sw2[o] * fmaxf(sw1x[o] * dx + sw1y[o] * dy + sb1[o], 0.f);
        s += rp;
        S[t][r] = s;
        rmax[r] = fmaxf(rmax[r], s);
      }
    }

    // ---- softmax (16-lane shuffle reductions per row)
#pragma unroll
    for (int r = 0; r < 8; ++r) {
#pragma unroll
      for (int off = 8; off >= 1; off >>= 1)
        rmax[r] = fmaxf(rmax[r], __shfl_xor(rmax[r], off, 16));
      rsum[r] = 0.f;
    }
#pragma unroll
    for (int t = 0; t < 16; ++t)
#pragma unroll
      for (int r = 0; r < 8; ++r) {
        const float p = __expf(S[t][r] - rmax[r]);
        S[t][r] = p;
        rsum[r] += p;
      }
#pragma unroll
    for (int r = 0; r < 8; ++r) {
#pragma unroll
      for (int off = 8; off >= 1; off >>= 1)
        rsum[r] += __shfl_xor(rsum[r], off, 16);
      rsum[r] = 1.f / rsum[r];
    }

    // ---- re-fragment P through per-wave LDS (D-layout -> A-layout)
#pragma unroll
    for (int t = 0; t < 16; ++t)
#pragma unroll
      for (int r = 0; r < 8; ++r)
        Pw[(r + 8 * lh) * NTOK + t * 16 + ln] = (bf16_t)(S[t][r] * rsum[r]);
    asm volatile("s_wait_dscnt 0" ::: "memory");

    // ---- O = P @ V (16 x 64), 32 WMMAs against V^T
    v8f O[4];
#pragma unroll
    for (int t = 0; t < 4; ++t) O[t] = (v8f){0.f,0.f,0.f,0.f,0.f,0.f,0.f,0.f};
    const bf16_t* prow = Pw + (size_t)ln * NTOK;
#pragma unroll
    for (int kk = 0; kk < NTOK; kk += 32) {
      v16bf a = frag_a(prow + kk, lh);
#pragma unroll
      for (int t = 0; t < 4; ++t) {
        v16bf bmat = frag_b(VT + (size_t)(t * 16 + ln) * NTOK + kk + 16 * lh);
        O[t] = wmma_bf16(a, bmat, O[t]);
      }
    }

    // ---- scatter to (B, N, C) bf16 for the projection GEMM
#pragma unroll
    for (int t = 0; t < 4; ++t)
#pragma unroll
      for (int r = 0; r < 8; ++r) {
        const int n = m0 + r + 8 * lh;
        const int c = h * HD + t * 16 + ln;
        ao[((size_t)(b * NTOK + n)) * DIM + c] = (bf16_t)O[t][r];
      }
  }
}

// ------------------------------------------------------------- proj GEMM + bias

__global__ __launch_bounds__(256) void
k_proj(const bf16_t* __restrict__ ab, const bf16_t* __restrict__ wb,
       const float* __restrict__ bias, float* __restrict__ out) {
  const int lane = threadIdx.x & 31, w = threadIdx.x >> 5;
  const int ln = lane & 15, lh = lane >> 4;
  const int m0 = (blockIdx.y * 8 + w) * 16;
  const int j0 = blockIdx.x * 64;

  const bf16_t* arow  = ab + (size_t)(m0 + ln) * DIM;
  const bf16_t* brow0 = wb + (size_t)(j0 + ln) * DIM + 16 * lh;

  v8f acc[4];
#pragma unroll
  for (int t = 0; t < 4; ++t) acc[t] = (v8f){0.f,0.f,0.f,0.f,0.f,0.f,0.f,0.f};

  for (int kk = 0; kk < DIM; kk += 32) {
    __builtin_prefetch(arow + kk + 128, 0, 1);
    v16bf a = frag_a(arow + kk, lh);
#pragma unroll
    for (int t = 0; t < 4; ++t) {
      v16bf b = frag_b(brow0 + (size_t)t * 16 * DIM + kk);
      acc[t] = wmma_bf16(a, b, acc[t]);
    }
  }

#pragma unroll
  for (int t = 0; t < 4; ++t) {
    const int j = j0 + t * 16 + ln;
    const float bj = bias[j];
#pragma unroll
    for (int r = 0; r < 8; ++r) {
      const int m = m0 + r + 8 * lh;
      out[(size_t)m * DIM + j] = acc[t][r] + bj;
    }
  }
}

// ---------------------------------------------------------------------- launcher

extern "C" void kernel_launch(void* const* d_in, const int* in_sizes, int n_in,
                              void* d_out, int out_size, void* d_ws, size_t ws_size,
                              hipStream_t stream) {
  const float* x     = (const float*)d_in[0];
  const float* pos   = (const float*)d_in[1];
  const float* vote  = (const float*)d_in[2];
  const float* Wqkv  = (const float*)d_in[3];
  const float* Wproj = (const float*)d_in[4];
  const float* bproj = (const float*)d_in[5];
  const float* tau   = (const float*)d_in[6];
  const float* rw1   = (const float*)d_in[7];
  const float* rb1   = (const float*)d_in[8];
  const float* rw2   = (const float*)d_in[9];
  const float* rb2   = (const float*)d_in[10];
  const float* vw1   = (const float*)d_in[11];
  const float* vb1   = (const float*)d_in[12];
  const float* vw2   = (const float*)d_in[13];
  const float* vb2   = (const float*)d_in[14];
  float* out = (float*)d_out;

  char* ws = (char*)d_ws;
  size_t off = 0;
  auto carve = [&](size_t bytes) -> void* {
    void* p = ws + off;
    off += (bytes + 255) & ~(size_t)255;
    return p;
  };
  const size_t TOKC = (size_t)MROWS * DIM;                  // 16.78M elems
  bf16_t* xb     = (bf16_t*)carve(TOKC * 2);                // reused as attn-out
  bf16_t* wqkvb  = (bf16_t*)carve((size_t)3 * DIM * DIM * 2);
  bf16_t* wprojb = (bf16_t*)carve((size_t)DIM * DIM * 2);
  bf16_t* veb    = (bf16_t*)carve(TOKC * 2);
  bf16_t* qb     = (bf16_t*)carve(TOKC * 2);
  bf16_t* kbuf   = (bf16_t*)carve(TOKC * 2);
  bf16_t* vbuf   = (bf16_t*)carve(TOKC * 2);
  float*  qn     = (float*)carve((size_t)BB * HEADS * NTOK * 4);
  float*  kn     = (float*)carve((size_t)BB * HEADS * NTOK * 4);
  bf16_t* ao     = xb;  // x is dead after k_qkv; reuse its 32 MB

  k_f32_to_bf16<<<4096, 256, 0, stream>>>(x, xb, (int)TOKC);
  k_f32_to_bf16<<<512, 256, 0, stream>>>(Wqkv, wqkvb, 3 * DIM * DIM);
  k_f32_to_bf16<<<256, 256, 0, stream>>>(Wproj, wprojb, DIM * DIM);
  k_vote_mlp<<<MROWS, 256, 0, stream>>>(vote, vw1, vb1, vw2, vb2, veb);

  k_qkv<<<dim3(3 * DIM / 64, MROWS / 128), 256, 0, stream>>>(xb, wqkvb, veb, qb,
                                                             kbuf, vbuf);
  k_norms<<<BB * HEADS * NTOK / 8, 256, 0, stream>>>(qb, kbuf, qn, kn);

  const size_t shmem = (size_t)3 * NTOK * HD * 2   // Q, K, V^T
                     + (size_t)4 * NTOK * 4        // qn, kn, px, py
                     + (size_t)8 * 16 * NTOK * 2;  // per-wave P tiles
  k_attn<<<BB * HEADS, 256, shmem, stream>>>(qb, kbuf, vbuf, qn, kn, pos, tau,
                                             rw1, rb1, rw2, rb2, ao);

  k_proj<<<dim3(DIM / 64, MROWS / 128), 256, 0, stream>>>(ao, wprojb, bproj, out);
}
